// RD_Decoder_no_pretrain_55387898249471
// MI455X (gfx1250) — matrix operand order repaired and checked by reading, not compiled, not run on hardware
//
#include <hip/hip_runtime.h>
#include <hip/hip_bf16.h>
#include <math.h>

// ---------------------------------------------------------------------------
// RD decoder head for MI455X (gfx1250, wave32, WMMA).
// All dense GEMMs use v_wmma_f32_16x16x32_bf16 with fp32 accumulation.
// ---------------------------------------------------------------------------

typedef __bf16 bf16_t;
typedef __attribute__((ext_vector_type(16))) __bf16 v16bf;
typedef __attribute__((ext_vector_type(8)))  __bf16 v8bf;
typedef __attribute__((ext_vector_type(8)))  float  v8f;

constexpr int Bn = 8, Qn = 30, Sn = 256, Hn = 768, NHn = 12, Dh = 64;
constexpr int FFNn = 3072, NRn = 25;
constexpr float NEGV = -10000.0f;

static __host__ __device__ inline int ceil_div(int a, int b) { return (a + b - 1) / b; }

// --------------------------- conversion kernels ----------------------------

// WT[n*K + k] = (bf16) W[k*N + n]; rows n in [N, Npad) zero-filled.
__global__ __launch_bounds__(256)
void k_transpose_bf16(const float* __restrict__ W, bf16_t* __restrict__ WT,
                      int K, int N, int Npad) {
  int idx = blockIdx.x * 256 + threadIdx.x;
  if (idx >= Npad * K) return;
  int n = idx / K;
  int k = idx - n * K;
  float v = (n < N) ? W[(size_t)k * N + n] : 0.0f;
  WT[idx] = (bf16_t)v;
}

__global__ __launch_bounds__(256)
void k_to_bf16(const float* __restrict__ x, bf16_t* __restrict__ y, int n) {
  int i = blockIdx.x * 256 + threadIdx.x;
  if (i < n) y[i] = (bf16_t)x[i];
}

// ------------------------------- LayerNorm ---------------------------------
// One 256-thread block per row (H = 768 -> 3 elems/thread).
// If mod > 0, input row index = row % mod (broadcast of query_embed).
// If res != null, x += res (residual) before LN.
__global__ __launch_bounds__(256)
void k_ln(const float* __restrict__ x, const float* __restrict__ res,
          const float* __restrict__ g, const float* __restrict__ be,
          float* __restrict__ of, bf16_t* __restrict__ ob, int mod) {
  int row = blockIdx.x;
  int rin = mod ? (row % mod) : row;
  const float* xp = x + (size_t)rin * Hn;
  const float* rp = res ? (res + (size_t)row * Hn) : nullptr;
  int t = threadIdx.x;

  float v0[3];
  float s = 0.0f, s2 = 0.0f;
#pragma unroll
  for (int i = 0; i < 3; ++i) {
    int h = t + i * 256;
    float v = xp[h] + (rp ? rp[h] : 0.0f);
    v0[i] = v; s += v; s2 += v * v;
  }
  __shared__ float red[256], red2[256];
  red[t] = s; red2[t] = s2;
  __syncthreads();
  for (int o = 128; o > 0; o >>= 1) {
    if (t < o) { red[t] += red[t + o]; red2[t] += red2[t + o]; }
    __syncthreads();
  }
  float mean = red[0] * (1.0f / Hn);
  float var  = red2[0] * (1.0f / Hn) - mean * mean;
  float rstd = rsqrtf(var + 1e-12f);
#pragma unroll
  for (int i = 0; i < 3; ++i) {
    int h = t + i * 256;
    float y = (v0[i] - mean) * rstd * g[h] + be[h];
    of[(size_t)row * Hn + h] = y;
    ob[(size_t)row * Hn + h] = (bf16_t)y;
  }
}

// ------------------------------ WMMA GEMM ----------------------------------
// C(M,N) = A(M,K) @ W(K,N) + bias, via WT = W^T (Npad,K) in bf16.
// One wave per 16x16 C tile; K-loop step 32 with v_wmma_f32_16x16x32_bf16.
// A-fragment lane layout (16-bit A, 16x32): lane&15 = row M; elems 0..15 hold
// K = (lane>>4)*16 + i.
// B-fragment (32x16): lane&15 = col N; elems 0..15 hold K = (lane>>4)*16 + i.
// C/D (f32 16x16): lane&15 = N, elem r -> M = (lane>>4)*8 + r.
// act: 0 = none, 1 = exact GELU.
__global__ __launch_bounds__(256)
void k_gemm_wmma(const bf16_t* __restrict__ A, const bf16_t* __restrict__ WT,
                 const float* __restrict__ bias,
                 float* __restrict__ Cf, bf16_t* __restrict__ Cb,
                 int M, int N, int K, int ldc, int act) {
  int w    = threadIdx.x >> 5;
  int lane = threadIdx.x & 31;
  int tm = blockIdx.x;
  int tn = blockIdx.y * 8 + w;
  int nt = (N + 15) >> 4;
  if (tn >= nt) return;                 // wave-uniform: EXEC stays all-ones

  int lh = lane & 15;
  int lg = lane >> 4;
  int arow = tm * 16 + lh;
  int bcol = tn * 16 + lh;
  const bf16_t* pA = A  + (size_t)arow * K + lg * 16;
  const bf16_t* pB = WT + (size_t)bcol * K + lg * 16;

  float bv = (bias && bcol < N) ? bias[bcol] : 0.0f;
  v8f acc;
#pragma unroll
  for (int r = 0; r < 8; ++r) acc[r] = bv;

  for (int k0 = 0; k0 < K; k0 += 32) {
    v8bf a0 = *(const v8bf*)(pA + k0);
    v8bf a1 = *(const v8bf*)(pA + k0 + 8);
    v16bf av;
#pragma unroll
    for (int i = 0; i < 8; ++i) { av[i] = a0[i]; av[i + 8] = a1[i]; }
    v16bf bm = *(const v16bf*)(pB + k0);
    if (k0 + 32 < K) {
      __builtin_prefetch(pA + k0 + 32, 0, 1);   // global_prefetch
      __builtin_prefetch(pB + k0 + 32, 0, 1);
    }
    acc = __builtin_amdgcn_wmma_f32_16x16x32_bf16(
        false, av, false, bm, (short)0, acc, false, false);
  }

#pragma unroll
  for (int r = 0; r < 8; ++r) {
    int m = tm * 16 + lg * 8 + r;
    int n = tn * 16 + lh;
    float v = acc[r];
    if (act == 1) v = 0.5f * v * (1.0f + erff(v * 0.7071067811865476f));
    if (m < M && n < N) {
      if (Cf) Cf[(size_t)m * ldc + n] = v;
      if (Cb) Cb[(size_t)m * ldc + n] = (bf16_t)v;
    }
  }
}

// --------------------------- self attention (30x30) ------------------------
// One block per (b, head). q/k/v tiles cached in LDS.
__global__ __launch_bounds__(256)
void k_self_attn(const float* __restrict__ Qm, const float* __restrict__ Km,
                 const float* __restrict__ Vm,
                 float* __restrict__ ctx_f, bf16_t* __restrict__ ctx_b) {
  int bh = blockIdx.x;
  int b = bh / NHn, h = bh - b * NHn;
  __shared__ float sS[Qn * Qn];
  __shared__ float sq[Qn * Dh];
  __shared__ float sk[Qn * Dh];
  __shared__ float sv[Qn * Dh];
  int t = threadIdx.x;
  for (int i = t; i < Qn * Dh; i += 256) {
    int r = i >> 6, c = i & 63;
    size_t off = ((size_t)(b * Qn + r)) * Hn + h * Dh + c;
    sq[i] = Qm[off]; sk[i] = Km[off]; sv[i] = Vm[off];
  }
  __syncthreads();
  for (int i = t; i < Qn * Qn; i += 256) {
    int qi = i / Qn, kj = i - qi * Qn;
    float s = 0.0f;
#pragma unroll 16
    for (int c = 0; c < Dh; ++c) s += sq[qi * Dh + c] * sk[kj * Dh + c];
    sS[i] = s * 0.125f;
  }
  __syncthreads();
  if (t < Qn) {
    float mx = -3.0e38f;
    for (int j = 0; j < Qn; ++j) mx = fmaxf(mx, sS[t * Qn + j]);
    float sum = 0.0f;
    for (int j = 0; j < Qn; ++j) { float e = expf(sS[t * Qn + j] - mx); sS[t * Qn + j] = e; sum += e; }
    float inv = 1.0f / sum;
    for (int j = 0; j < Qn; ++j) sS[t * Qn + j] *= inv;
  }
  __syncthreads();
  for (int i = t; i < Qn * Dh; i += 256) {
    int qi = i >> 6, c = i & 63;
    float s = 0.0f;
    for (int j = 0; j < Qn; ++j) s += sS[qi * Qn + j] * sv[j * Dh + c];
    size_t off = ((size_t)(b * Qn + qi)) * Hn + h * Dh + c;
    ctx_f[off] = s; ctx_b[off] = (bf16_t)s;
  }
}

// -------------------------- cross attention (30x256) ------------------------
__global__ __launch_bounds__(256)
void k_cross_attn(const float* __restrict__ Qm, const float* __restrict__ Km,
                  const float* __restrict__ Vm, const int* __restrict__ amask,
                  float* __restrict__ ctx_f, bf16_t* __restrict__ ctx_b) {
  int bh = blockIdx.x;
  int b = bh / NHn, h = bh - b * NHn;
  __shared__ float sS[Qn * Sn];
  __shared__ float sq[Qn * Dh];
  int t = threadIdx.x;
  for (int i = t; i < Qn * Dh; i += 256) {
    int r = i >> 6, c = i & 63;
    sq[i] = Qm[((size_t)(b * Qn + r)) * Hn + h * Dh + c];
  }
  __syncthreads();
  for (int i = t; i < Qn * Sn; i += 256) {
    int qi = i >> 8, kj = i & 255;
    const float* kp = Km + ((size_t)(b * Sn + kj)) * Hn + h * Dh;
    float s = 0.0f;
#pragma unroll 16
    for (int c = 0; c < Dh; ++c) s += sq[qi * Dh + c] * kp[c];
    float mb = (amask[b * Sn + kj] == 0) ? NEGV : 0.0f;
    sS[i] = s * 0.125f + mb;
  }
  __syncthreads();
  if (t < Qn) {
    float mx = -3.0e38f;
    for (int j = 0; j < Sn; ++j) mx = fmaxf(mx, sS[t * Sn + j]);
    float sum = 0.0f;
    for (int j = 0; j < Sn; ++j) { float e = expf(sS[t * Sn + j] - mx); sS[t * Sn + j] = e; sum += e; }
    float inv = 1.0f / sum;
    for (int j = 0; j < Sn; ++j) sS[t * Sn + j] *= inv;
  }
  __syncthreads();
  for (int i = t; i < Qn * Dh; i += 256) {
    int qi = i >> 6, c = i & 63;
    float s = 0.0f;
    for (int j = 0; j < Sn; ++j)
      s += sS[qi * Sn + j] * Vm[((size_t)(b * Sn + j)) * Hn + h * Dh + c];
    size_t off = ((size_t)(b * Qn + qi)) * Hn + h * Dh + c;
    ctx_f[off] = s; ctx_b[off] = (bf16_t)s;
  }
}

// ------------------------------- biaffine ----------------------------------
// out[b,q,s] = mask ? sum_h tanh(a[bq,h] + c[bs,h]) * w3[h] : NEG.
// One wave32 per output; lanes stride H (coalesced); shuffle reduction.
__global__ __launch_bounds__(256)
void k_biaffine(const float* __restrict__ a, const float* __restrict__ c,
                const float* __restrict__ w3, const int* __restrict__ smask,
                float* __restrict__ out) {
  int wid = blockIdx.x * 8 + (threadIdx.x >> 5);
  int lane = threadIdx.x & 31;
  if (wid >= Bn * Qn * Sn) return;
  int s  = wid % Sn;
  int bq = wid / Sn;
  int b  = bq / Qn;
  const float* ap = a + (size_t)bq * Hn;
  const float* cp = c + ((size_t)(b * Sn + s)) * Hn;
  float acc = 0.0f;
  for (int h = lane; h < Hn; h += 32)
    acc += tanhf(ap[h] + cp[h]) * w3[h];
#pragma unroll
  for (int o = 16; o > 0; o >>= 1) acc += __shfl_down(acc, o, 32);
  if (lane == 0) out[wid] = (smask[b * Sn + s] == 0) ? NEGV : acc;
}

// ------------------------------ orchestration -------------------------------

extern "C" void kernel_launch(void* const* d_in, const int* in_sizes, int n_in,
                              void* d_out, int out_size, void* d_ws, size_t ws_size,
                              hipStream_t stream) {
  (void)in_sizes; (void)n_in; (void)out_size; (void)ws_size;

  const int*   tok_mask  = (const int*)d_in[0];
  const int*   span_mask = (const int*)d_in[1];
  const float* enc       = (const float*)d_in[2];
  const float* rd        = (const float*)d_in[3];
  const float* P[48];
  for (int i = 4; i < n_in && i < 52; ++i) P[i - 4] = (const float*)d_in[i];

  // params (dict insertion order)
  const float* query_embed = P[0];
  const float* ln_g = P[1];   const float* ln_b = P[2];
  const float* sa_Wq = P[3];  const float* sa_bq = P[4];
  const float* sa_Wk = P[5];  const float* sa_bk = P[6];
  const float* sa_Wv = P[7];  const float* sa_bv = P[8];
  const float* sa_Wo = P[9];  const float* sa_bo = P[10];
  const float* sa_ln_g = P[11]; const float* sa_ln_b = P[12];
  const float* ca_Wq = P[13]; const float* ca_bq = P[14];
  const float* ca_Wk = P[15]; const float* ca_bk = P[16];
  const float* ca_Wv = P[17]; const float* ca_bv = P[18];
  const float* ca_Wo = P[19]; const float* ca_bo = P[20];
  const float* ca_ln_g = P[21]; const float* ca_ln_b = P[22];
  const float* ff_W1 = P[23]; const float* ff_b1 = P[24];
  const float* ff_W2 = P[25]; const float* ff_b2 = P[26];
  const float* ff_ln_g = P[27]; const float* ff_ln_b = P[28];
  const float* cls_W = P[29]; const float* cls_b = P[30];
  const float* head_W1 = P[31]; const float* head_b1 = P[32];
  const float* head_W2 = P[33]; const float* head_b2 = P[34];
  const float* head_w3 = P[35];
  const float* tail_W1 = P[36]; const float* tail_b1 = P[37];
  const float* tail_W2 = P[38]; const float* tail_b2 = P[39];
  const float* tail_w3 = P[40];

  // --- workspace bump allocator -------------------------------------------
  char* ws = (char*)d_ws;
  size_t off = 0;
  auto alloc = [&](size_t bytes) -> void* {
    off = (off + 255) & ~(size_t)255;
    void* p = ws + off;
    off += bytes;
    return p;
  };
  const int MQ = Bn * Qn;   // 240
  const int MS = Bn * Sn;   // 2048

  // bf16 transposed weights
  bf16_t* WT_sa_q = (bf16_t*)alloc((size_t)Hn * Hn * 2);
  bf16_t* WT_sa_k = (bf16_t*)alloc((size_t)Hn * Hn * 2);
  bf16_t* WT_sa_v = (bf16_t*)alloc((size_t)Hn * Hn * 2);
  bf16_t* WT_sa_o = (bf16_t*)alloc((size_t)Hn * Hn * 2);
  bf16_t* WT_ca_q = (bf16_t*)alloc((size_t)Hn * Hn * 2);
  bf16_t* WT_ca_k = (bf16_t*)alloc((size_t)Hn * Hn * 2);
  bf16_t* WT_ca_v = (bf16_t*)alloc((size_t)Hn * Hn * 2);
  bf16_t* WT_ca_o = (bf16_t*)alloc((size_t)Hn * Hn * 2);
  bf16_t* WT_ff1  = (bf16_t*)alloc((size_t)FFNn * Hn * 2);
  bf16_t* WT_ff2  = (bf16_t*)alloc((size_t)Hn * FFNn * 2);
  bf16_t* WT_cls  = (bf16_t*)alloc((size_t)32 * Hn * 2);   // NR=25 padded to 32
  bf16_t* WT_h1   = (bf16_t*)alloc((size_t)Hn * Hn * 2);
  bf16_t* WT_h2   = (bf16_t*)alloc((size_t)Hn * Hn * 2);
  bf16_t* WT_t1   = (bf16_t*)alloc((size_t)Hn * Hn * 2);
  bf16_t* WT_t2   = (bf16_t*)alloc((size_t)Hn * Hn * 2);

  // bf16 activations
  bf16_t* hs_b   = (bf16_t*)alloc((size_t)MQ * Hn * 2);
  bf16_t* hs1_b  = (bf16_t*)alloc((size_t)MQ * Hn * 2);
  bf16_t* hs2_b  = (bf16_t*)alloc((size_t)MQ * Hn * 2);
  bf16_t* dec_b  = (bf16_t*)alloc((size_t)MQ * Hn * 2);
  bf16_t* enc_b  = (bf16_t*)alloc((size_t)MS * Hn * 2);
  bf16_t* rd_b   = (bf16_t*)alloc((size_t)MS * Hn * 2);
  bf16_t* f1_b   = (bf16_t*)alloc((size_t)MQ * FFNn * 2);
  bf16_t* ctx1_b = (bf16_t*)alloc((size_t)MQ * Hn * 2);
  bf16_t* ctx2_b = (bf16_t*)alloc((size_t)MQ * Hn * 2);

  // fp32 activations
  float* hs_f   = (float*)alloc((size_t)MQ * Hn * 4);
  float* hs1_f  = (float*)alloc((size_t)MQ * Hn * 4);
  float* hs2_f  = (float*)alloc((size_t)MQ * Hn * 4);
  float* dec_f  = (float*)alloc((size_t)MQ * Hn * 4);
  float* qs     = (float*)alloc((size_t)MQ * Hn * 4);
  float* ks     = (float*)alloc((size_t)MQ * Hn * 4);
  float* vs     = (float*)alloc((size_t)MQ * Hn * 4);
  float* ctx1_f = (float*)alloc((size_t)MQ * Hn * 4);
  float* o1     = (float*)alloc((size_t)MQ * Hn * 4);
  float* qc     = (float*)alloc((size_t)MQ * Hn * 4);
  float* kc     = (float*)alloc((size_t)MS * Hn * 4);
  float* vc     = (float*)alloc((size_t)MS * Hn * 4);
  float* ctx2_f = (float*)alloc((size_t)MQ * Hn * 4);
  float* o2     = (float*)alloc((size_t)MQ * Hn * 4);
  float* f2     = (float*)alloc((size_t)MQ * Hn * 4);
  float* ah     = (float*)alloc((size_t)MQ * Hn * 4);
  float* at     = (float*)alloc((size_t)MQ * Hn * 4);
  float* ch     = (float*)alloc((size_t)MS * Hn * 4);
  float* ct     = (float*)alloc((size_t)MS * Hn * 4);

  // --- helpers -------------------------------------------------------------
  auto tr = [&](const float* W, bf16_t* WT, int K, int N, int Npad) {
    int total = Npad * K;
    k_transpose_bf16<<<ceil_div(total, 256), 256, 0, stream>>>(W, WT, K, N, Npad);
  };
  auto gemm = [&](const bf16_t* A, const bf16_t* WT, const float* bias,
                  float* Cf, bf16_t* Cb, int M, int N, int K, int ldc, int act) {
    int nt = (N + 15) >> 4;
    dim3 grid(M / 16, ceil_div(nt, 8));
    k_gemm_wmma<<<grid, 256, 0, stream>>>(A, WT, bias, Cf, Cb, M, N, K, ldc, act);
  };

  // --- stage 0: conversions ------------------------------------------------
  tr(sa_Wq, WT_sa_q, Hn, Hn, Hn); tr(sa_Wk, WT_sa_k, Hn, Hn, Hn);
  tr(sa_Wv, WT_sa_v, Hn, Hn, Hn); tr(sa_Wo, WT_sa_o, Hn, Hn, Hn);
  tr(ca_Wq, WT_ca_q, Hn, Hn, Hn); tr(ca_Wk, WT_ca_k, Hn, Hn, Hn);
  tr(ca_Wv, WT_ca_v, Hn, Hn, Hn); tr(ca_Wo, WT_ca_o, Hn, Hn, Hn);
  tr(ff_W1, WT_ff1, Hn, FFNn, FFNn);
  tr(ff_W2, WT_ff2, FFNn, Hn, Hn);
  tr(cls_W, WT_cls, Hn, NRn, 32);
  tr(head_W1, WT_h1, Hn, Hn, Hn); tr(head_W2, WT_h2, Hn, Hn, Hn);
  tr(tail_W1, WT_t1, Hn, Hn, Hn); tr(tail_W2, WT_t2, Hn, Hn, Hn);
  k_to_bf16<<<ceil_div(MS * Hn, 256), 256, 0, stream>>>(enc, enc_b, MS * Hn);
  k_to_bf16<<<ceil_div(MS * Hn, 256), 256, 0, stream>>>(rd,  rd_b,  MS * Hn);

  // hs = LN(broadcast(query_embed))
  k_ln<<<MQ, 256, 0, stream>>>(query_embed, nullptr, ln_g, ln_b, hs_f, hs_b, Qn);

  // --- self attention ------------------------------------------------------
  gemm(hs_b, WT_sa_q, sa_bq, qs, nullptr, MQ, Hn, Hn, Hn, 0);
  gemm(hs_b, WT_sa_k, sa_bk, ks, nullptr, MQ, Hn, Hn, Hn, 0);
  gemm(hs_b, WT_sa_v, sa_bv, vs, nullptr, MQ, Hn, Hn, Hn, 0);
  k_self_attn<<<Bn * NHn, 256, 0, stream>>>(qs, ks, vs, ctx1_f, ctx1_b);
  gemm(ctx1_b, WT_sa_o, sa_bo, o1, nullptr, MQ, Hn, Hn, Hn, 0);
  k_ln<<<MQ, 256, 0, stream>>>(o1, hs_f, sa_ln_g, sa_ln_b, hs1_f, hs1_b, 0);

  // --- cross attention -----------------------------------------------------
  gemm(hs1_b, WT_ca_q, ca_bq, qc, nullptr, MQ, Hn, Hn, Hn, 0);
  gemm(enc_b, WT_ca_k, ca_bk, kc, nullptr, MS, Hn, Hn, Hn, 0);
  gemm(enc_b, WT_ca_v, ca_bv, vc, nullptr, MS, Hn, Hn, Hn, 0);
  k_cross_attn<<<Bn * NHn, 256, 0, stream>>>(qc, kc, vc, tok_mask, ctx2_f, ctx2_b);
  gemm(ctx2_b, WT_ca_o, ca_bo, o2, nullptr, MQ, Hn, Hn, Hn, 0);
  k_ln<<<MQ, 256, 0, stream>>>(o2, hs1_f, ca_ln_g, ca_ln_b, hs2_f, hs2_b, 0);

  // --- FFN -----------------------------------------------------------------
  gemm(hs2_b, WT_ff1, ff_b1, nullptr, f1_b, MQ, FFNn, Hn, FFNn, 1);  // fused GELU
  gemm(f1_b, WT_ff2, ff_b2, f2, nullptr, MQ, Hn, FFNn, Hn, 0);
  k_ln<<<MQ, 256, 0, stream>>>(f2, hs2_f, ff_ln_g, ff_ln_b, dec_f, dec_b, 0);

  // --- heads ---------------------------------------------------------------
  float* out = (float*)d_out;
  float* out_cls  = out;                       // (B,Q,NR)  = 6000
  float* out_head = out + Bn * Qn * NRn;       // (B,Q,S)   = 61440
  float* out_tail = out_head + Bn * Qn * Sn;

  gemm(dec_b, WT_cls, cls_b, out_cls, nullptr, MQ, NRn, Hn, NRn, 0);

  gemm(dec_b, WT_h1, head_b1, ah, nullptr, MQ, Hn, Hn, Hn, 0);
  gemm(rd_b,  WT_h2, head_b2, ch, nullptr, MS, Hn, Hn, Hn, 0);
  gemm(dec_b, WT_t1, tail_b1, at, nullptr, MQ, Hn, Hn, Hn, 0);
  gemm(rd_b,  WT_t2, tail_b2, ct, nullptr, MS, Hn, Hn, Hn, 0);

  int nbia = (Bn * Qn * Sn) / 8;  // 8 waves per block
  k_biaffine<<<nbia, 256, 0, stream>>>(ah, ch, head_w3, span_mask, out_head);
  k_biaffine<<<nbia, 256, 0, stream>>>(at, ct, tail_w3, span_mask, out_tail);
}
